// ChemGeomFeatEncoder_53171695124881
// MI455X (gfx1250) — compile-verified, matrix operand order checked by reading, not run on hardware
//
#include <hip/hip_runtime.h>
#include <hip/hip_bf16.h>
#include <math.h>

// ---------------------------------------------------------------------------
// ChemGeomFeatEncoder for MI455X (gfx1250, wave32, WMMA 16x16x32 f16/f32-acc)
//
// BatchNorms folded into Linears by a prep kernel that also transposes the
// weights into [N][Kpad] f16 so WMMA B-fragments are two contiguous
// global_load_b128 per lane. A-fragments come from wave-private LDS tiles
// (two ds_load_b128 per lane). No cross-wave LDS sharing -> no barriers.
// Activations use hardware approx ops (v_rcp_f32 / v_exp_f32 / v_log_f32)
// to keep the VALU epilogue short against the WMMA pipe.
// ---------------------------------------------------------------------------

typedef __attribute__((ext_vector_type(16))) _Float16 v16h;
typedef __attribute__((ext_vector_type(8)))  _Float16 v8h;
typedef __attribute__((ext_vector_type(8)))  float    v8f;

#define WMMA_F16(a, b, c) \
  __builtin_amdgcn_wmma_f32_16x16x32_f16(false, (a), false, (b), (short)0, (c), false, false)

#define NV   65536
#define NE   1048576
#define AS   72    // LDS row stride (halves) for 64-wide tiles  (144B = 9*16B)
#define HS   136   // LDS row stride for 128-wide tiles          (272B = 17*16B)
#define XS   200   // LDS row stride for 192-wide concat tiles   (400B = 25*16B)

// --- fragment loaders ------------------------------------------------------
// 16-bit A/B fragment, matrix stored row-major with given half-stride.
// Lane l (0..15): row = row0 + l, K-halves {koff..koff+7, 16+koff..16+koff+7}
// with koff = (l>>4)*8 (mirrored for lanes 16..31). Matches ISA 7.12.2.
template <typename P>
__device__ inline v16h load_frag(P base, int row0, int k0, int stride, int lane) {
  int r    = row0 + (lane & 15);
  int koff = (lane >> 4) * 8;
  const _Float16* p = (const _Float16*)base + (size_t)r * stride + k0;
  v8h lo = *(const v8h*)(p + koff);
  v8h hi = *(const v8h*)(p + 16 + koff);
  v16h out;
#pragma unroll
  for (int i = 0; i < 8; ++i) { out[i] = lo[i]; out[i + 8] = hi[i]; }
  return out;
}

// --- fast activations (hardware approx: v_rcp_f32, v_exp_f32, v_log_f32) ---
__device__ inline float fast_rcp(float x)  { return __builtin_amdgcn_rcpf(x); }
__device__ inline float sigm_f(float x)    { return fast_rcp(1.0f + __expf(-x)); }
__device__ inline float silu_f(float x)    { return x * sigm_f(x); }
__device__ inline float softplus_f(float x) {
  // max(x,0) + log(1 + exp(-|x|)); __logf -> v_log_f32
  return fmaxf(x, 0.0f) + __logf(1.0f + __expf(-fabsf(x)));
}

// ---------------------------------------------------------------------------
// Prep: fold BN into Linear, transpose to [N][Kpad] f16, emit folded bias f32
// ---------------------------------------------------------------------------
__global__ void fold_transpose(const float* __restrict__ W,   // [K][N]
                               const float* __restrict__ b,   // [N]
                               const float* __restrict__ bn,  // [4][N] g,be,m,v
                               _Float16* __restrict__ Wt,     // [N][Kpad]
                               float* __restrict__ bout,      // [N]
                               int K, int N, int Kpad) {
  int idx = blockIdx.x * blockDim.x + threadIdx.x;
  if (idx >= N * Kpad) return;
  int n = idx / Kpad, k = idx - n * Kpad;
  float g  = bn[0 * N + n];
  float be = bn[1 * N + n];
  float m  = bn[2 * N + n];
  float v  = bn[3 * N + n];
  float s  = g * rsqrtf(v + 1e-5f);
  float w  = (k < K) ? W[(size_t)k * N + n] * s : 0.0f;
  Wt[(size_t)n * Kpad + k] = (_Float16)w;
  if (k == 0) bout[n] = (b[n] - m) * s + be;
}

// ---------------------------------------------------------------------------
// Chem MLP + gated scatter-add.  256 thr = 8 waves, each wave owns 16 edges.
// ---------------------------------------------------------------------------
__global__ __launch_bounds__(256) void chem_kernel(
    const float* __restrict__ chem, const int* __restrict__ nbr,
    const _Float16* __restrict__ W1t, const float* __restrict__ b1,
    const _Float16* __restrict__ W2t, const float* __restrict__ b2,
    float* __restrict__ hchem) {
  __shared__ __align__(16) _Float16 ldsA[8 * 16 * AS];
  __shared__ __align__(16) _Float16 ldsH[8 * 16 * HS];
  const int lane = threadIdx.x & 31;
  const int wave = threadIdx.x >> 5;
  const int e0   = blockIdx.x * 128 + wave * 16;
  const int nn   = lane & 15, mh = lane >> 4;

  _Float16* A  = ldsA + wave * 16 * AS;
  _Float16* Hb = ldsH + wave * 16 * HS;

  if (lane == 0) { __builtin_prefetch(W1t, 0, 0); __builtin_prefetch(W2t, 0, 0); }

  // stage chem tile [16 x 34] f32 -> f16 LDS, zero-padded K -> 64
  const float* src = chem + (size_t)e0 * 34;
  for (int idx = lane; idx < 16 * 64; idx += 32) {
    int r = idx >> 6, k = idx & 63;
    A[r * AS + k] = (_Float16)((k < 34) ? src[r * 34 + k] : 0.0f);
  }

  // layer 1: H = SiLU(A @ W1' + b1'), N = 128
  for (int nt = 0; nt < 8; ++nt) {
    v8f acc = {};
#pragma unroll
    for (int ks = 0; ks < 2; ++ks) {
      v16h a = load_frag(A,   0,       ks * 32, AS, lane);
      v16h b = load_frag(W1t, nt * 16, ks * 32, 64, lane);
      acc = WMMA_F16(a, b, acc);
    }
    float bb = b1[nt * 16 + nn];
    _Float16* dst = Hb + (mh * 8) * HS + nt * 16 + nn;
#pragma unroll
    for (int r = 0; r < 8; ++r) dst[r * HS] = (_Float16)silu_f(acc[r] + bb);
  }

  // layer 2: [filter | core] = H @ W2' + b2'; gate; segment-sum scatter
  for (int nt = 0; nt < 8; ++nt) {
    v8f accF = {}, accC = {};
#pragma unroll
    for (int ks = 0; ks < 4; ++ks) {
      v16h a  = load_frag(Hb,  0,             ks * 32, HS,  lane);
      v16h bf = load_frag(W2t, nt * 16,       ks * 32, 128, lane);
      v16h bc = load_frag(W2t, 128 + nt * 16, ks * 32, 128, lane);
      accF = WMMA_F16(a, bf, accF);
      accC = WMMA_F16(a, bc, accC);
    }
    float bF  = b2[nt * 16 + nn];
    float bC  = b2[128 + nt * 16 + nn];
    int   col = nt * 16 + nn;
#pragma unroll
    for (int r = 0; r < 8; ++r) {
      float gated = sigm_f(accF[r] + bF) * softplus_f(accC[r] + bC);
      int vid = nbr[e0 + mh * 8 + r];
      unsafeAtomicAdd(hchem + (size_t)vid * 128 + col, gated);  // global_atomic_add_f32
    }
  }
}

// ---------------------------------------------------------------------------
// Geom MLP + concat + feat MLP.  128 thr = 4 waves, each wave 16 vertices.
// ---------------------------------------------------------------------------
__global__ __launch_bounds__(128) void geomfeat_kernel(
    const float* __restrict__ geom, const float* __restrict__ hchem,
    const _Float16* __restrict__ Wg1t, const float* __restrict__ bg1,
    const _Float16* __restrict__ Wg2t, const float* __restrict__ bg2,
    const _Float16* __restrict__ Wf1t, const float* __restrict__ bf1,
    const _Float16* __restrict__ Wf2t, const float* __restrict__ bf2,
    float* __restrict__ out) {
  __shared__ __align__(16) _Float16 ldsAg[4 * 16 * AS];
  __shared__ __align__(16) _Float16 ldsG1[4 * 16 * AS];
  __shared__ __align__(16) _Float16 ldsX [4 * 16 * XS];  // [hchem(128) | hgeom(64)]
  __shared__ __align__(16) _Float16 ldsY [4 * 16 * HS];
  const int lane = threadIdx.x & 31;
  const int wave = threadIdx.x >> 5;
  const int v0   = blockIdx.x * 64 + wave * 16;
  const int nn   = lane & 15, mh = lane >> 4;

  _Float16* Ag = ldsAg + wave * 16 * AS;
  _Float16* G1 = ldsG1 + wave * 16 * AS;
  _Float16* X  = ldsX  + wave * 16 * XS;
  _Float16* Y  = ldsY  + wave * 16 * HS;

  // stage geom tile [16 x 48] -> f16, pad K -> 64
  const float* gsrc = geom + (size_t)v0 * 48;
  for (int idx = lane; idx < 16 * 64; idx += 32) {
    int r = idx >> 6, k = idx & 63;
    Ag[r * AS + k] = (_Float16)((k < 48) ? gsrc[r * 48 + k] : 0.0f);
  }
  // stage h_chem tile [16 x 128] f32 -> f16 into X cols 0..127
  const float* csrc = hchem + (size_t)v0 * 128;
  for (int idx = lane; idx < 16 * 128; idx += 32) {
    int r = idx >> 7, k = idx & 127;
    X[r * XS + k] = (_Float16)csrc[r * 128 + k];
  }

  // g1 = SiLU(Ag @ Wg1' + bg1'), N = 64
  for (int nt = 0; nt < 4; ++nt) {
    v8f acc = {};
#pragma unroll
    for (int ks = 0; ks < 2; ++ks) {
      v16h a = load_frag(Ag,   0,       ks * 32, AS, lane);
      v16h b = load_frag(Wg1t, nt * 16, ks * 32, 64, lane);
      acc = WMMA_F16(a, b, acc);
    }
    float bb = bg1[nt * 16 + nn];
    _Float16* dst = G1 + (mh * 8) * AS + nt * 16 + nn;
#pragma unroll
    for (int r = 0; r < 8; ++r) dst[r * AS] = (_Float16)silu_f(acc[r] + bb);
  }

  // h_geom = g1 @ Wg2' + bg2' -> X cols 128..191
  for (int nt = 0; nt < 4; ++nt) {
    v8f acc = {};
#pragma unroll
    for (int ks = 0; ks < 2; ++ks) {
      v16h a = load_frag(G1,   0,       ks * 32, AS, lane);
      v16h b = load_frag(Wg2t, nt * 16, ks * 32, 64, lane);
      acc = WMMA_F16(a, b, acc);
    }
    float bb = bg2[nt * 16 + nn];
    _Float16* dst = X + (mh * 8) * XS + 128 + nt * 16 + nn;
#pragma unroll
    for (int r = 0; r < 8; ++r) dst[r * XS] = (_Float16)(acc[r] + bb);
  }

  // y = SiLU(X @ Wf1' + bf1'), K = 192, N = 128
  for (int nt = 0; nt < 8; ++nt) {
    v8f acc = {};
#pragma unroll
    for (int ks = 0; ks < 6; ++ks) {
      v16h a = load_frag(X,    0,       ks * 32, XS,  lane);
      v16h b = load_frag(Wf1t, nt * 16, ks * 32, 192, lane);
      acc = WMMA_F16(a, b, acc);
    }
    float bb = bf1[nt * 16 + nn];
    _Float16* dst = Y + (mh * 8) * HS + nt * 16 + nn;
#pragma unroll
    for (int r = 0; r < 8; ++r) dst[r * HS] = (_Float16)silu_f(acc[r] + bb);
  }

  // out = Y @ Wf2' + bf2', K = 128, N = 128 (f32 output)
  for (int nt = 0; nt < 8; ++nt) {
    v8f acc = {};
#pragma unroll
    for (int ks = 0; ks < 4; ++ks) {
      v16h a = load_frag(Y,    0,       ks * 32, HS,  lane);
      v16h b = load_frag(Wf2t, nt * 16, ks * 32, 128, lane);
      acc = WMMA_F16(a, b, acc);
    }
    float bb = bf2[nt * 16 + nn];
    float* dst = out + (size_t)(v0 + mh * 8) * 128 + nt * 16 + nn;
#pragma unroll
    for (int r = 0; r < 8; ++r) dst[(size_t)r * 128] = acc[r] + bb;
  }
}

// ---------------------------------------------------------------------------
// Host launcher
// ---------------------------------------------------------------------------
extern "C" void kernel_launch(void* const* d_in, const int* in_sizes, int n_in,
                              void* d_out, int out_size, void* d_ws, size_t ws_size,
                              hipStream_t stream) {
  const float* chem = (const float*)d_in[0];
  const float* geom = (const float*)d_in[1];
  const int*   nbr  = (const int*)  d_in[2];
  const float* w1  = (const float*)d_in[3];  const float* b1  = (const float*)d_in[4];  const float* bn1  = (const float*)d_in[5];
  const float* w2  = (const float*)d_in[6];  const float* b2  = (const float*)d_in[7];  const float* bn2  = (const float*)d_in[8];
  const float* wg1 = (const float*)d_in[9];  const float* bg1 = (const float*)d_in[10]; const float* bng1 = (const float*)d_in[11];
  const float* wg2 = (const float*)d_in[12]; const float* bg2 = (const float*)d_in[13]; const float* bng2 = (const float*)d_in[14];
  const float* wf1 = (const float*)d_in[15]; const float* bf1 = (const float*)d_in[16]; const float* bnf1 = (const float*)d_in[17];
  const float* wf2 = (const float*)d_in[18]; const float* bf2 = (const float*)d_in[19]; const float* bnf2 = (const float*)d_in[20];

  char* ws = (char*)d_ws;
  size_t off = 0;
  float*    hchem = (float*)(ws + off);          off += (size_t)NV * 128 * 4;   // 32 MB
  _Float16* W1t   = (_Float16*)(ws + off);       off += 128 * 64  * 2;
  _Float16* W2t   = (_Float16*)(ws + off);       off += 256 * 128 * 2;
  _Float16* Wg1t  = (_Float16*)(ws + off);       off += 64  * 64  * 2;
  _Float16* Wg2t  = (_Float16*)(ws + off);       off += 64  * 64  * 2;
  _Float16* Wf1t  = (_Float16*)(ws + off);       off += 128 * 192 * 2;
  _Float16* Wf2t  = (_Float16*)(ws + off);       off += 128 * 128 * 2;
  float* B1  = (float*)(ws + off); off += 128 * 4;
  float* B2  = (float*)(ws + off); off += 256 * 4;
  float* BG1 = (float*)(ws + off); off += 64 * 4;
  float* BG2 = (float*)(ws + off); off += 64 * 4;
  float* BF1 = (float*)(ws + off); off += 128 * 4;
  float* BF2 = (float*)(ws + off); off += 128 * 4;

  hipMemsetAsync(hchem, 0, (size_t)NV * 128 * 4, stream);

  auto prep = [&](const float* W, const float* b, const float* bn,
                  _Float16* Wt, float* bo, int K, int N, int Kpad) {
    int total = N * Kpad;
    fold_transpose<<<(total + 255) / 256, 256, 0, stream>>>(W, b, bn, Wt, bo, K, N, Kpad);
  };
  prep(w1,  b1,  bn1,  W1t,  B1,  34,  128, 64);
  prep(w2,  b2,  bn2,  W2t,  B2,  128, 256, 128);
  prep(wg1, bg1, bng1, Wg1t, BG1, 48,  64,  64);
  prep(wg2, bg2, bng2, Wg2t, BG2, 64,  64,  64);
  prep(wf1, bf1, bnf1, Wf1t, BF1, 192, 128, 192);
  prep(wf2, bf2, bnf2, Wf2t, BF2, 128, 128, 128);

  chem_kernel<<<NE / 128, 256, 0, stream>>>(chem, nbr, W1t, B1, W2t, B2, hchem);

  geomfeat_kernel<<<NV / 64, 128, 0, stream>>>(geom, hchem,
                                               Wg1t, BG1, Wg2t, BG2,
                                               Wf1t, BF1, Wf2t, BF2,
                                               (float*)d_out);
}
